// MultHeadAttention_87479893885545
// MI455X (gfx1250) — matrix operand order repaired and checked by reading, not compiled, not run on hardware
//
#include <hip/hip_runtime.h>

// ---------------------------------------------------------------------------
// MultiHeadAttention forward on gfx1250 (CDNA5).
// B=4, T=1024, E=1024, H=16, D=64.
// GEMMs: 64x64 block tile / 4-wave WG, LDS double-buffered, A-tile staged via
// global_load_async_to_lds_b128 (ASYNCcnt), B-tile staged via the Tensor Data
// Mover (TENSORcnt).  Attention: wave-level causal flash with bf16 WMMA.
// ---------------------------------------------------------------------------

#define BB 4
#define TT 1024
#define EE 1024
#define HH 16
#define DD 64

typedef __attribute__((ext_vector_type(16))) __bf16 v16bf;
typedef __attribute__((ext_vector_type(8)))  float  v8f;
typedef __attribute__((ext_vector_type(4)))  unsigned int u32x4;
typedef __attribute__((ext_vector_type(8)))  int i32x8;
typedef __attribute__((ext_vector_type(4)))  int i32x4;

union Frag16 {           // 16 bf16 values = 8 dwords (A or B operand of 16x16x32)
    v16bf v;
    unsigned int u[8];
};

__device__ __forceinline__ unsigned short f32_to_bf16(float f) {
    union { float f; unsigned int u; } c; c.f = f;
    unsigned int r = c.u + 0x7FFFu + ((c.u >> 16) & 1u);  // round-to-nearest-even
    return (unsigned short)(r >> 16);
}

__device__ __forceinline__ v8f wmma_bf16(const Frag16& a, const Frag16& b, v8f c) {
    return __builtin_amdgcn_wmma_f32_16x16x32_bf16(
        false, a.v, false, b.v, (short)0, c, false, false);
}

// A-matrix 16x32 bf16 fragment from row-major [rows, ld] (contiguous k = 16h + s).
__device__ __forceinline__ void load_a(Frag16& f, const unsigned short* p,
                                       int ld, int row0, int k0) {
    int lane = threadIdx.x & 31;
    const unsigned short* base = p + (size_t)(row0 + (lane & 15)) * ld + k0 + ((lane >> 4) << 4);
#pragma unroll
    for (int j = 0; j < 4; ++j)
        f.u[j] = *(const unsigned int*)(base + 2 * j);
#pragma unroll
    for (int j = 0; j < 4; ++j)
        f.u[4 + j] = *(const unsigned int*)(base + 8 + 2 * j);
}

// B-matrix 32x16 fragment where memory holds B^T row-major: [N, K].
__device__ __forceinline__ void load_b_nt(Frag16& f, const unsigned short* p,
                                          int ld, int col0, int k0) {
    int lane = threadIdx.x & 31;
    const unsigned short* base = p + (size_t)(col0 + (lane & 15)) * ld + k0 + ((lane >> 4) << 4);
#pragma unroll
    for (int j = 0; j < 8; ++j)
        f.u[j] = *(const unsigned int*)(base + 2 * j);
}

// B-matrix 32x16 fragment where memory holds B row-major: [K, N].
__device__ __forceinline__ void load_b_nn(Frag16& f, const unsigned short* p,
                                          int ld, int k0, int col0) {
    int lane = threadIdx.x & 31;
    const unsigned short* base = p + (size_t)(k0 + ((lane >> 4) << 4)) * ld + col0 + (lane & 15);
#pragma unroll
    for (int j = 0; j < 8; ++j) {
        unsigned int lo = base[(2 * j) * ld];
        unsigned int hi = base[(2 * j + 1) * ld];
        f.u[j] = lo | (hi << 16);
    }
}

__device__ __forceinline__ unsigned lds_offset(const void* p) {
    // low 32 bits of the flat address of a __shared__ object are the LDS offset
    return (unsigned)(unsigned long long)p;
}

// Issue one async global->LDS b128 copy (ASYNCcnt path).
__device__ __forceinline__ void async_b128(unsigned lds_off, const void* gaddr) {
    unsigned long long ga = (unsigned long long)gaddr;
    asm volatile("global_load_async_to_lds_b128 %0, %1, off"
                 :: "v"(lds_off), "v"(ga) : "memory");
}

__device__ __forceinline__ void wait_async0() {
    asm volatile("s_wait_asynccnt 0x0" ::: "memory");
}

// TDM load of a [rows=64, cols=32] bf16 tile from a [EE, EE] row-major tensor
// into LDS (row-major 64x32).  D# per CDNA5 ISA.
__device__ __forceinline__ void tdm_load_tile_64x32(const unsigned short* tensor,
                                                    int row0, int k0,
                                                    unsigned lds_off) {
    unsigned long long ga =
        (unsigned long long)(const void*)(tensor + (size_t)row0 * EE + k0);
    u32x4 g0;
    g0[0] = 1u;                                   // count=1 (valid), load, user
    g0[1] = lds_off;                              // LDS byte address
    g0[2] = (unsigned)(ga & 0xFFFFFFFFu);         // global_addr[31:0]
    g0[3] = (unsigned)((ga >> 32) & 0x1FFFFFFu)   // global_addr[56:32]
          | (2u << 30);                           // type = 2 ("image")
    i32x8 g1;
    g1[0] = (int)(1u << 16);                      // data_size = 1 -> 2 bytes
    g1[1] = (int)((EE & 0xFFFFu) << 16);          // tensor_dim0[15:0]
    g1[2] = (int)((EE & 0xFFFFu) << 16);          // dim0 hi=0 | tensor_dim1[15:0]
    g1[3] = (int)(32u << 16);                     // dim1 hi=0 | tile_dim0 = 32 (k)
    g1[4] = (int)64;                              // tile_dim1 = 64 rows, tile_dim2 = 0
    g1[5] = (int)EE;                              // tensor_dim0_stride[31:0]
    g1[6] = 0;                                    // stride hi | dim1_stride lo
    g1[7] = 0;
    i32x4 z4 = (i32x4)0;
#if defined(__clang_major__) && (__clang_major__ >= 23)
    i32x8 z8 = (i32x8)0;
    __builtin_amdgcn_tensor_load_to_lds(g0, g1, z4, z4, z8, 0);
#else
    __builtin_amdgcn_tensor_load_to_lds(g0, g1, z4, z4, 0);
#endif
}

// ---------------------------------------------------------------------------
__global__ void cvt_bf16_kernel(const float* __restrict__ s,
                                unsigned short* __restrict__ d, int n) {
    int i = blockIdx.x * blockDim.x + threadIdx.x;
    if (i < n) d[i] = f32_to_bf16(s[i]);
}

// ---------------------------------------------------------------------------
// NT GEMM: C[m,n] = sum_k A[m,k] * W[n,k], M x N x K with K = EE.
// Block = 128 threads (4 waves), tile 64(M) x 64(N) x 32(K), LDS double-buffer.
// EPI=0: scatter bf16 into [B,H,T,D] (blockIdx.z selects q/k/v matrix).
// EPI=1: f32 output + bias.
// ---------------------------------------------------------------------------
template <int EPI>
__global__ __launch_bounds__(128)
void gemm64_kernel(const unsigned short* __restrict__ A,
                   const unsigned short* __restrict__ W,
                   const float* __restrict__ bias,
                   unsigned short* __restrict__ out_bf,
                   float* __restrict__ out_f) {
    const int m0 = blockIdx.x * 64;
    const int n0 = blockIdx.y * 64;
    const unsigned short* Wp = W + (size_t)blockIdx.z * EE * EE;

    __shared__ unsigned short As[2][64 * 32];   // 4 KB each
    __shared__ unsigned short Bs[2][64 * 32];

    const int tid  = threadIdx.x;
    const int wave = tid >> 5;
    const int lane = tid & 31;

    const unsigned ldsA[2] = { lds_offset(&As[0][0]), lds_offset(&As[1][0]) };
    const unsigned ldsB[2] = { lds_offset(&Bs[0][0]), lds_offset(&Bs[1][0]) };

    // stage tile kt into buffer (kt & 1): A via async-to-LDS, B via TDM (wave 0)
    auto issue_tile = [&](int kt) {
        const int buf = kt & 1;
        const int k0  = kt * 32;
        // A tile: 64 rows x 32 bf16 = 256 b128 transfers, 2 per thread
#pragma unroll
        for (int s = 0; s < 2; ++s) {
            const int i = tid + s * 128;          // b128 index, row = i/4, seg = i%4
            const unsigned short* g =
                A + (size_t)(m0 + (i >> 2)) * EE + k0 + (i & 3) * 8;
            async_b128(ldsA[buf] + (unsigned)i * 16u, g);
        }
        if (wave == 0)
            tdm_load_tile_64x32(Wp, n0, k0, ldsB[buf]);
    };

    issue_tile(0);

    v8f acc[4];
#pragma unroll
    for (int nb = 0; nb < 4; ++nb) acc[nb] = {};

    const int NT = EE / 32;
    for (int kt = 0; kt < NT; ++kt) {
        wait_async0();
        if (wave == 0) __builtin_amdgcn_s_wait_tensorcnt(0);
        __syncthreads();                          // tile kt resident in LDS
        if (kt + 1 < NT) issue_tile(kt + 1);      // overlap copy with compute

        const int buf = kt & 1;
        const unsigned short* ap = &As[buf][0];
        const unsigned short* bp = &Bs[buf][0];
        Frag16 af;
        load_a(af, ap, 32, wave * 16, 0);
#pragma unroll
        for (int nb = 0; nb < 4; ++nb) {
            Frag16 bf;
            load_b_nt(bf, bp, 32, nb * 16, 0);
            acc[nb] = wmma_bf16(af, bf, acc[nb]);
        }
    }

    const int ncol = lane & 15;
    const int mhi  = (lane >> 4) << 3;
#pragma unroll
    for (int nb = 0; nb < 4; ++nb) {
#pragma unroll
        for (int r = 0; r < 8; ++r) {
            const int m = m0 + wave * 16 + mhi + r;
            const int n = n0 + nb * 16 + ncol;
            if (EPI == 0) {
                // scatter to [B,H,T,D] bf16 ; m = b*T+t, n = h*D+d
                unsigned short* dst =
                    out_bf + (size_t)blockIdx.z * BB * HH * TT * DD;
                const int bidx = m / TT, t = m % TT;
                const int h = n / DD, d = n % DD;
                dst[(((size_t)bidx * HH + h) * TT + t) * DD + d] =
                    f32_to_bf16(acc[nb][r]);
            } else {
                out_f[(size_t)m * EE + n] = acc[nb][r] + bias[n];
            }
        }
    }
}

// ---------------------------------------------------------------------------
// Causal flash attention: one wave per (16 query rows, head, batch).
// ---------------------------------------------------------------------------
__global__ __launch_bounds__(32)
void attn_kernel(const unsigned short* __restrict__ q,
                 const unsigned short* __restrict__ k,
                 const unsigned short* __restrict__ v,
                 unsigned short* __restrict__ o) {
    const int t0 = blockIdx.x * 16;
    const int bh = blockIdx.y;
    const int b = bh / HH, h = bh % HH;
    const int lane = threadIdx.x & 31;
    const int ncol = lane & 15;
    const int mhi  = (lane >> 4) << 3;

    const unsigned short* qb = q + (size_t)bh * TT * DD;
    const unsigned short* kb = k + (size_t)bh * TT * DD;
    const unsigned short* vb = v + (size_t)bh * TT * DD;

    Frag16 qa0, qa1;
    load_a(qa0, qb, DD, t0, 0);
    load_a(qa1, qb, DD, t0, 32);

    v8f oacc[4];
    float mi[8], li[8];
#pragma unroll
    for (int nb = 0; nb < 4; ++nb) oacc[nb] = {};
#pragma unroll
    for (int r = 0; r < 8; ++r) { mi[r] = -1e30f; li[r] = 0.0f; }

    __shared__ unsigned short Ptile[16 * 32];

    const float scale = 0.125f;                  // 1/sqrt(64)
    const int nblk = (t0 + 15) / 32 + 1;

    for (int blk = 0; blk < nblk; ++blk) {
        const int s0 = blk * 32;

        v8f sf[2];
        Frag16 kf;
#pragma unroll
        for (int half = 0; half < 2; ++half) {
            sf[half] = {};
            load_b_nt(kf, kb, DD, s0 + 16 * half, 0);
            sf[half] = wmma_bf16(qa0, kf, sf[half]);
            load_b_nt(kf, kb, DD, s0 + 16 * half, 32);
            sf[half] = wmma_bf16(qa1, kf, sf[half]);
        }

#pragma unroll
        for (int half = 0; half < 2; ++half) {
            const int sg = s0 + 16 * half + ncol;
#pragma unroll
            for (int r = 0; r < 8; ++r) {
                const int tg = t0 + mhi + r;
                float val = sf[half][r] * scale;
                sf[half][r] = (sg > tg) ? -1e30f : val;
            }
        }

#pragma unroll
        for (int r = 0; r < 8; ++r) {
            float rm = fmaxf(sf[0][r], sf[1][r]);
#pragma unroll
            for (int xm = 1; xm < 16; xm <<= 1)
                rm = fmaxf(rm, __shfl_xor(rm, xm, 32));
            const float mnew  = fmaxf(mi[r], rm);
            const float alpha = __expf(mi[r] - mnew);
            const float p0 = __expf(sf[0][r] - mnew);
            const float p1 = __expf(sf[1][r] - mnew);
            float rs = p0 + p1;
#pragma unroll
            for (int xm = 1; xm < 16; xm <<= 1)
                rs += __shfl_xor(rs, xm, 32);
            li[r] = li[r] * alpha + rs;
            mi[r] = mnew;
#pragma unroll
            for (int nb = 0; nb < 4; ++nb) oacc[nb][r] *= alpha;

            const int row = mhi + r;
            Ptile[row * 32 + ncol]      = f32_to_bf16(p0);
            Ptile[row * 32 + 16 + ncol] = f32_to_bf16(p1);
        }
        __syncthreads();

        Frag16 pf, vf;
        load_a(pf, Ptile, 32, 0, 0);
#pragma unroll
        for (int nb = 0; nb < 4; ++nb) {
            load_b_nn(vf, vb, DD, s0, nb * 16);
            oacc[nb] = wmma_bf16(pf, vf, oacc[nb]);
        }
        __syncthreads();
    }

#pragma unroll
    for (int r = 0; r < 8; ++r) {
        const float inv = 1.0f / li[r];
        const int row = t0 + mhi + r;
        const size_t base = ((size_t)b * TT + row) * EE + h * DD;
#pragma unroll
        for (int nb = 0; nb < 4; ++nb)
            o[base + nb * 16 + ncol] = f32_to_bf16(oacc[nb][r] * inv);
    }
}

// ---------------------------------------------------------------------------
extern "C" void kernel_launch(void* const* d_in, const int* in_sizes, int n_in,
                              void* d_out, int out_size, void* d_ws, size_t ws_size,
                              hipStream_t stream) {
    const float* x  = (const float*)d_in[0];   // [B,T,E]
    const float* Wq = (const float*)d_in[1];   // [H,D,E]
    const float* Wk = (const float*)d_in[2];
    const float* Wv = (const float*)d_in[3];
    const float* Wp = (const float*)d_in[4];   // [E,E]
    const float* bp = (const float*)d_in[5];   // [E]
    float* out = (float*)d_out;                // [B,T,E]

    const size_t nX = (size_t)BB * TT * EE;    // 4M
    const size_t nW = (size_t)EE * EE;         // 1M (H*D == E)
    unsigned short* ws   = (unsigned short*)d_ws;
    unsigned short* xb   = ws;                 // x bf16
    unsigned short* w3   = xb + nX;            // Wq|Wk|Wv bf16
    unsigned short* wpb  = w3 + 3 * nW;        // Wp bf16
    unsigned short* qkv  = wpb + nW;           // q|k|v bf16 [B,H,T,D] each
    unsigned short* obuf = qkv + 3 * nX;       // attention output bf16 [B,T,E]

    const int cvtB = 256;
    cvt_bf16_kernel<<<(int)((nX + cvtB - 1) / cvtB), cvtB, 0, stream>>>(x, xb, (int)nX);
    cvt_bf16_kernel<<<(int)((nW + cvtB - 1) / cvtB), cvtB, 0, stream>>>(Wq, w3 + 0 * nW, (int)nW);
    cvt_bf16_kernel<<<(int)((nW + cvtB - 1) / cvtB), cvtB, 0, stream>>>(Wk, w3 + 1 * nW, (int)nW);
    cvt_bf16_kernel<<<(int)((nW + cvtB - 1) / cvtB), cvtB, 0, stream>>>(Wv, w3 + 2 * nW, (int)nW);
    cvt_bf16_kernel<<<(int)((nW + cvtB - 1) / cvtB), cvtB, 0, stream>>>(Wp, wpb, (int)nW);

    // QKV projections: M=4096, N=1024 per matrix, K=1024
    gemm64_kernel<0><<<dim3(BB * TT / 64, EE / 64, 3), 128, 0, stream>>>(
        xb, w3, nullptr, qkv, nullptr);

    // Causal attention
    attn_kernel<<<dim3(TT / 16, BB * HH), 32, 0, stream>>>(
        qkv, qkv + nX, qkv + 2 * nX, obuf);

    // Output projection + bias
    gemm64_kernel<1><<<dim3(BB * TT / 64, EE / 64, 1), 128, 0, stream>>>(
        obuf, wpb, bp, nullptr, out);
}
